// SpatialCodec_Model_34256659153061
// MI455X (gfx1250) — compile-verified, matrix operand order checked by reading, not run on hardware
//
#include <hip/hip_runtime.h>

// ---------------------------------------------------------------------------
// SpatialCodec forward for MI455X (gfx1250, wave32).
// STFT / ISTFT are done as GEMMs against (i)DFT matrices using
// v_wmma_f32_16x16x32_f16 with an f16 hi/lo split (3 WMMAs per k-tile) for
// ~f32 accuracy.  The memory-bound deep-filter einsum streams rtf_g with
// coalesced, NON-TEMPORAL float2 loads (146 MB single-use) and k_cov writes
// feats (57 MB single-use) non-temporally, keeping the reused 13 MB spectra
// and DFT tables resident in the 192 MB L2.
// ---------------------------------------------------------------------------

typedef _Float16 v16h __attribute__((ext_vector_type(16)));
typedef _Float16 v8h  __attribute__((ext_vector_type(8)));
typedef _Float16 v2h  __attribute__((ext_vector_type(2)));
typedef float    v8f  __attribute__((ext_vector_type(8)));
typedef float    v2f  __attribute__((ext_vector_type(2)));

#define NFFT  640
#define HOP   320
#define NF    321            // freq bins
#define NT    301            // frames
#define TLEN  96000
#define NSIG1 17             // 16 x-channels (B=2,C=8) + ref_g batch0
#define M1R   (NSIG1*NT)     // 5117 real rows
#define M1P   5120           // padded to mult of 16
#define N1P   656            // 642 (re/im of 321 bins) padded to mult of 16
#define NCOLS 642
#define M2R   (7*NT)         // 2107
#define M2P   2112
#define K2P   672            // 642 padded to mult of 32
#define PI_F  3.14159265358979323846f

// ---------------------------- init kernels ---------------------------------

__global__ void k_init_win(float* __restrict__ win) {
  int n = blockIdx.x * blockDim.x + threadIdx.x;
  if (n < NFFT) win[n] = 0.5f * (1.0f - cosf(2.0f * PI_F * (float)n / (float)NFFT));
}

// Forward DFT matrix, transposed (N x K layout = Dt[col][sample]) and split
// into f16 hi/lo.  col 2f -> cos(2*pi*f*n/N), col 2f+1 -> -sin(...).
__global__ void k_init_dt(_Float16* __restrict__ Dh, _Float16* __restrict__ Dl) {
  int idx = blockIdx.x * blockDim.x + threadIdx.x;
  if (idx >= N1P * NFFT) return;
  int n = idx / NFFT, k = idx % NFFT;
  float v = 0.0f;
  if (n < NCOLS) {
    int f  = n >> 1;
    int ph = (f * k) % NFFT;                       // exact phase reduction
    float ang = 2.0f * PI_F * (float)ph / (float)NFFT;
    v = (n & 1) ? -sinf(ang) : cosf(ang);
  }
  _Float16 h = (_Float16)v;
  Dh[idx] = h;
  Dl[idx] = (_Float16)(v - (float)h);
}

// Inverse (real) DFT matrix, transposed (N x K = Et[sample][coef]).
// coef 2f -> c_f*cos/N, 2f+1 -> -c_f*sin/N, c_f = 2 except bins 0 and N/2.
__global__ void k_init_et(_Float16* __restrict__ Eh, _Float16* __restrict__ El) {
  int idx = blockIdx.x * blockDim.x + threadIdx.x;
  if (idx >= NFFT * K2P) return;
  int n = idx / K2P, k2 = idx % K2P;
  float v = 0.0f;
  if (k2 < NCOLS) {
    int f = k2 >> 1;
    float c = (f == 0 || f == NF - 1) ? 1.0f : 2.0f;
    int ph = (f * n) % NFFT;
    float ang = 2.0f * PI_F * (float)ph / (float)NFFT;
    v = ((k2 & 1) ? -c * sinf(ang) : c * cosf(ang)) / (float)NFFT;
  }
  _Float16 h = (_Float16)v;
  Eh[idx] = h;
  El[idx] = (_Float16)(v - (float)h);
}

// Windowed, zero-padded frames for all 17 signals, row-major [M1P][640], f16 hi/lo.
__global__ void k_frames(const float* __restrict__ x, const float* __restrict__ refg,
                         const float* __restrict__ win,
                         _Float16* __restrict__ Ah, _Float16* __restrict__ Al) {
  int idx = blockIdx.x * blockDim.x + threadIdx.x;
  if (idx >= M1P * NFFT) return;
  int m = idx / NFFT, n = idx % NFFT;
  int s = m / NT, t = m % NT;
  float v = 0.0f;
  if (s < NSIG1) {
    int q = t * HOP + n - HOP;                    // center=True zero pad
    if (q >= 0 && q < TLEN) {
      float src = (s < 16) ? x[s * TLEN + q] : refg[q];
      v = src * win[n];
    }
  }
  _Float16 h = (_Float16)v;
  Ah[idx] = h;
  Al[idx] = (_Float16)(v - (float)h);
}

// ---------------------------- WMMA GEMM ------------------------------------
// One wave computes one 16x16 f32 tile.  A is MxK row-major, B is NxK
// row-major (i.e. transposed operand).  Per §7.12.2, lane l (m=l&15, h=l>>4)
// needs k = {k0+8h..+7} and {k0+16+8h..+7}: two contiguous b128 loads.

__device__ __forceinline__ v16h frag_ld(const _Float16* p) {
  v8h a = *(const v8h*)(p);
  v8h b = *(const v8h*)(p + 16);
  v16h r;
#pragma unroll
  for (int e = 0; e < 8; ++e) { r[e] = a[e]; r[e + 8] = b[e]; }
  return r;
}

__global__ void k_gemm_wmma(const _Float16* __restrict__ Ah, const _Float16* __restrict__ Al,
                            const _Float16* __restrict__ Bh, const _Float16* __restrict__ Bl,
                            float* __restrict__ C, const float* __restrict__ col_scale,
                            int Mt, int Nt, int K, int lda, int ldb, int ldc) {
  int wave = (int)((blockIdx.x * blockDim.x + threadIdx.x) >> 5);
  int lane = (int)(threadIdx.x & 31);
  int tm = wave / Nt, tn = wave % Nt;
  if (tm >= Mt) return;                           // wave-uniform exit (EXEC stays all-1)
  int r = lane & 15, h = lane >> 4;

  const _Float16* pAh = Ah + (size_t)(tm * 16 + r) * lda;
  const _Float16* pAl = Al + (size_t)(tm * 16 + r) * lda;
  const _Float16* pBh = Bh + (size_t)(tn * 16 + r) * ldb;
  const _Float16* pBl = Bl + (size_t)(tn * 16 + r) * ldb;

  v8f acc = {};
  for (int k0 = 0; k0 < K; k0 += 32) {
    int off = k0 + 8 * h;
    v16h ah = frag_ld(pAh + off);
    v16h al = frag_ld(pAl + off);
    v16h bh = frag_ld(pBh + off);
    v16h bl = frag_ld(pBl + off);
    // f32 ~= (ah+al)*(bh+bl): keep hi*hi + hi*lo + lo*hi (lo*lo below f32 ulp)
    acc = __builtin_amdgcn_wmma_f32_16x16x32_f16(false, ah, false, bh, (short)0, acc, false, false);
    acc = __builtin_amdgcn_wmma_f32_16x16x32_f16(false, ah, false, bl, (short)0, acc, false, false);
    acc = __builtin_amdgcn_wmma_f32_16x16x32_f16(false, al, false, bh, (short)0, acc, false, false);
  }
  int n = tn * 16 + r;
  float s = col_scale ? col_scale[n] : 1.0f;      // synthesis window on ISTFT pass
  float* pc = C + (size_t)(tm * 16 + 8 * h) * ldc + n;
#pragma unroll
  for (int rr = 0; rr < 8; ++rr) pc[rr * ldc] = acc[rr] * s;
}

// ---------------------------- covariance features --------------------------
// feats[b, 2p]   = Re(S_i * conj(S_j)),  feats[b, 2p+1] = Im(...)
// feats[b, 72/73] = Re/Im of S[REF_CH=3].   SG rows: (b*8+c)*NT + t, cols 2f.
// feats is written once and never re-read: store non-temporally.
__global__ void k_cov(const float* __restrict__ SG, float* __restrict__ feats) {
  int idx = blockIdx.x * blockDim.x + threadIdx.x;
  if (idx >= 2 * 37 * NF * NT) return;
  int ft = idx % (NF * NT);
  int pc = (idx / (NF * NT)) % 37;
  int b  = idx / (NF * NT * 37);
  int t = ft % NT, f = ft / NT;
  if (pc < 36) {
    int i = 0, rem = pc;                          // triu_indices(8) row-major
    while (rem >= 8 - i) { rem -= 8 - i; ++i; }
    int j = i + rem;
    float2 Si = *(const float2*)(SG + ((size_t)((b * 8 + i) * NT + t)) * N1P + 2 * f);
    float2 Sj = *(const float2*)(SG + ((size_t)((b * 8 + j) * NT + t)) * N1P + 2 * f);
    float cr = Si.x * Sj.x + Si.y * Sj.y;         // S_i * conj(S_j)
    float ci = Si.y * Sj.x - Si.x * Sj.y;
    __builtin_nontemporal_store(cr, feats + ((size_t)(b * 74 + 2 * pc)     * NF + f) * NT + t);
    __builtin_nontemporal_store(ci, feats + ((size_t)(b * 74 + 2 * pc + 1) * NF + f) * NT + t);
  } else {
    float2 Sr = *(const float2*)(SG + ((size_t)((b * 8 + 3) * NT + t)) * N1P + 2 * f);
    __builtin_nontemporal_store(Sr.x, feats + ((size_t)(b * 74 + 72) * NF + f) * NT + t);
    __builtin_nontemporal_store(Sr.y, feats + ((size_t)(b * 74 + 73) * NF + f) * NT + t);
  }
}

// ---------------------------- deep filtering -------------------------------
// est[m,f,t] = sum_{ki,kj} rtf_c[0,m,ki,kj,f,t] * refS[f+ki-1, t+kj-4]
// Output written directly as Est GEMM operand rows (f16 hi/lo), row = m*NT+t,
// cols 2f / 2f+1 = Re/Im.  Only batch 0 is observable in the output.
// rtf is a 146 MB single-use stream: load non-temporally so the refS spectra
// (reused 27x) stay L2-resident.
__global__ void k_deepfilter(const float* __restrict__ rtf, const float* __restrict__ SG,
                             _Float16* __restrict__ Eh, _Float16* __restrict__ El) {
  int idx = blockIdx.x * blockDim.x + threadIdx.x;
  if (idx >= 7 * NF * NT) return;
  int t = idx % NT;
  int f = (idx / NT) % NF;
  int m = idx / (NT * NF);
  float er = 0.0f, ei = 0.0f;
#pragma unroll
  for (int ki = 0; ki < 3; ++ki) {
    int fi = f + ki - 1;
    bool fok = (fi >= 0) && (fi < NF);
#pragma unroll
    for (int kj = 0; kj < 9; ++kj) {
      int tj = t + kj - 4;
      if (fok && tj >= 0 && tj < NT) {
        float2 p = *(const float2*)(SG + ((size_t)(16 * NT + tj)) * N1P + 2 * fi);
        v2f w = __builtin_nontemporal_load(
            (const v2f*)(rtf + 2 * ((size_t)(((m * 3 + ki) * 9 + kj) * NF + f) * NT + t)));
        er += w[0] * p.x - w[1] * p.y;
        ei += w[0] * p.y + w[1] * p.x;
      }
    }
  }
  int row = m * NT + t;
  _Float16 hr = (_Float16)er, hi = (_Float16)ei;
  v2h hs; hs[0] = hr; hs[1] = hi;
  v2h ls; ls[0] = (_Float16)(er - (float)hr); ls[1] = (_Float16)(ei - (float)hi);
  *(v2h*)(Eh + (size_t)row * K2P + 2 * f) = hs;
  *(v2h*)(El + (size_t)row * K2P + 2 * f) = ls;
}

// ---------------------------- overlap-add + interleave ---------------------
// Exactly two windowed frames contribute per sample; wsum >= 0.5 always.
// out layout is all_ch[0].T  ->  out[j*8 + ch].
__global__ void k_out(const float* __restrict__ fr2, const float* __restrict__ win,
                      const float* __restrict__ refg, float* __restrict__ out0) {
  int idx = blockIdx.x * blockDim.x + threadIdx.x;
  if (idx >= 8 * TLEN) return;
  int j = idx % TLEN, ch = idx / TLEN;
  float v;
  if (ch == 3) {
    v = refg[j];                                  // REF_CH passthrough (batch 0)
  } else {
    int sig = (ch < 3) ? ch : ch - 1;
    int t0 = j / HOP + 1;                         // tau = j + 320
    int n0 = j % HOP;
    float a = fr2[(size_t)(sig * NT + t0 - 1) * NFFT + n0 + HOP];
    float b = fr2[(size_t)(sig * NT + t0)     * NFFT + n0];
    float w0 = win[n0], w1 = win[n0 + HOP];
    float ws = w0 * w0 + w1 * w1;
    v = (a + b) / (ws > 1e-11f ? ws : 1.0f);
  }
  __builtin_nontemporal_store(v, out0 + (size_t)j * 8 + ch);
}

// ---------------------------- host launcher --------------------------------

extern "C" void kernel_launch(void* const* d_in, const int* in_sizes, int n_in,
                              void* d_out, int out_size, void* d_ws, size_t ws_size,
                              hipStream_t stream) {
  (void)in_sizes; (void)n_in; (void)out_size; (void)ws_size;
  const float* x    = (const float*)d_in[0];      // [2,8,96000]
  const float* rtf  = (const float*)d_in[1];      // [2,7,3,9,321,301,2]
  const float* refg = (const float*)d_in[2];      // [2,1,96000]
  float* out = (float*)d_out;                     // [96000*8] ++ feats[2,74,321,301]

  char* ws = (char*)d_ws;
  size_t o = 0;
  auto take = [&](size_t bytes) -> char* {
    char* p = ws + o; o = (o + bytes + 255) & ~(size_t)255; return p;
  };
  _Float16* FrAh = (_Float16*)take((size_t)M1P * NFFT * 2 * 2);   // hi ++ lo
  _Float16* FrAl = FrAh + (size_t)M1P * NFFT;
  _Float16* Dth  = (_Float16*)take((size_t)N1P * NFFT * 2 * 2);
  _Float16* Dtl  = Dth + (size_t)N1P * NFFT;
  float*    SG   = (float*)take((size_t)M1P * N1P * 4);
  float*    win  = (float*)take(NFFT * 4);
  _Float16* Eth  = (_Float16*)take((size_t)NFFT * K2P * 2 * 2);
  _Float16* Etl  = Eth + (size_t)NFFT * K2P;
  _Float16* EsAh = (_Float16*)take((size_t)M2P * K2P * 2 * 2);
  _Float16* EsAl = EsAh + (size_t)M2P * K2P;
  float*    Fr2  = (float*)take((size_t)M2P * NFFT * 4);

  const int B = 256;
  k_init_win<<<(NFFT + B - 1) / B, B, 0, stream>>>(win);
  k_init_dt <<<(N1P * NFFT + B - 1) / B, B, 0, stream>>>(Dth, Dtl);
  k_init_et <<<(NFFT * K2P + B - 1) / B, B, 0, stream>>>(Eth, Etl);
  k_frames  <<<(M1P * NFFT + B - 1) / B, B, 0, stream>>>(x, refg, win, FrAh, FrAl);

  {
    int tiles = (M1P / 16) * (N1P / 16);
    int blocks = (tiles + (B / 32) - 1) / (B / 32);
    k_gemm_wmma<<<blocks, B, 0, stream>>>(FrAh, FrAl, Dth, Dtl, SG, nullptr,
                                          M1P / 16, N1P / 16, NFFT, NFFT, NFFT, N1P);
  }

  k_cov<<<(2 * 37 * NF * NT + B - 1) / B, B, 0, stream>>>(SG, out + (size_t)8 * TLEN);

  hipMemsetAsync(EsAh, 0, (size_t)M2P * K2P * 2 * 2, stream);     // pad rows/cols
  k_deepfilter<<<(7 * NF * NT + B - 1) / B, B, 0, stream>>>(rtf, SG, EsAh, EsAl);

  {
    int tiles = (M2P / 16) * (NFFT / 16);
    int blocks = (tiles + (B / 32) - 1) / (B / 32);
    k_gemm_wmma<<<blocks, B, 0, stream>>>(EsAh, EsAl, Eth, Etl, Fr2, win,
                                          M2P / 16, NFFT / 16, K2P, K2P, K2P, NFFT);
  }

  k_out<<<(8 * TLEN + B - 1) / B, B, 0, stream>>>(Fr2, win, refg, out);
}
